// SGN_31885837206089
// MI455X (gfx1250) — matrix operand order repaired and checked by reading, not compiled, not run on hardware
//
#include <hip/hip_runtime.h>

// ---------------------------------------------------------------------------
// Types
// ---------------------------------------------------------------------------
typedef __bf16 bf16_t;
typedef bf16_t v16bf __attribute__((ext_vector_type(16)));
typedef bf16_t v8bf  __attribute__((ext_vector_type(8)));
typedef float  v8f   __attribute__((ext_vector_type(8)));

#define N_NODES 100000
#define N_EDGES 500000
// d_out layout: h_e [E,128] | h_v2 [N,64] | u [64]
#define OUT_HE   0
#define OUT_HV2  64000000
#define OUT_U    70400000

// ws layout (float index for f32 region, then byte offsets)
#define AGG_F    0                       // [N,128] f32 (atomic scatter target)
#define CNT_F    12800000                // [N]     f32
#define SUMHE_F  12900000                // [128]   f32
#define SUMHV_F  12900128                // [128]   f32
#define UTMP_F   12900256                // [64]    f32
#define ZCOUNT   12900320                // floats to zero
#define HV_B     51601280                // [N,128] bf16  (25,600,000 B)
#define WEB_B    77201280                // 8*9*512  bf16 frag (73,728 B)
#define WNB_B    77275008                // 8*10*512 bf16 frag (81,920 B)
#define WA_B     77356928                // 8*10*512 bf16 frag (81,920 B)
#define WB_B     77438848                // 4*4*512  bf16 frag (16,384 B)

static __device__ __forceinline__ v8f wmma_bf16(v16bf a, v16bf b, v8f c) {
  return __builtin_amdgcn_wmma_f32_16x16x32_bf16(false, a, false, b,
                                                 (short)0, c, false, false);
}

// ---------------------------------------------------------------------------
// Utility kernels
// ---------------------------------------------------------------------------
__global__ void zero_f32(float* p, int n) {
  for (int i = blockIdx.x * blockDim.x + threadIdx.x; i < n;
       i += gridDim.x * blockDim.x)
    p[i] = 0.f;
}

// Pack W[K][Nn] (f32, row-major) into WMMA-B bf16 fragment layout:
// frag f = t*ksteps + s ; within frag: lane l, half h ->
//   n = t*16 + (l&15), k = s*32 + (l>>4)*8 + (h&7) + (h>>3)*16
__global__ void pack_w(const float* __restrict__ W, bf16_t* __restrict__ dst,
                       int ksteps, int ntiles, int Nn) {
  int total = ntiles * ksteps * 512;
  for (int i = blockIdx.x * blockDim.x + threadIdx.x; i < total;
       i += gridDim.x * blockDim.x) {
    int h = i & 15;
    int l = (i >> 4) & 31;
    int f = i >> 9;
    int s = f % ksteps;
    int t = f / ksteps;
    int n = t * 16 + (l & 15);
    int k = s * 32 + (l >> 4) * 8 + (h & 7) + (h >> 3) * 16;
    dst[i] = (bf16_t)W[(size_t)k * Nn + n];
  }
}

// ---------------------------------------------------------------------------
// A-fragment row loaders (8 contiguous f32, boundaries are multiples of 8)
// ---------------------------------------------------------------------------
__device__ __forceinline__ void load8_edge(const float* ce, const float* cx,
                                           const float* g, int k0, float* d) {
  const float* src;
  if (k0 < 96)        src = ce + k0;          // cat_e row
  else if (k0 < 224)  src = cx + (k0 - 96);   // cat_x[sender] row
  else                src = g + (k0 - 224);   // global_attr
  float4 a = *(const float4*)src;
  float4 b = *(const float4*)(src + 4);
  d[0]=a.x; d[1]=a.y; d[2]=a.z; d[3]=a.w;
  d[4]=b.x; d[5]=b.y; d[6]=b.z; d[7]=b.w;
}

__device__ __forceinline__ void load8_node(const float* ag, const float* cx,
                                           const float* g, float inv, int k0,
                                           float* d) {
  if (k0 < 128) {
    float4 a = *(const float4*)(ag + k0);
    float4 b = *(const float4*)(ag + k0 + 4);
    d[0]=a.x*inv; d[1]=a.y*inv; d[2]=a.z*inv; d[3]=a.w*inv;
    d[4]=b.x*inv; d[5]=b.y*inv; d[6]=b.z*inv; d[7]=b.w*inv;
  } else {
    const float* src = (k0 < 256) ? (cx + (k0 - 128)) : (g + (k0 - 256));
    float4 a = *(const float4*)src;
    float4 b = *(const float4*)(src + 4);
    d[0]=a.x; d[1]=a.y; d[2]=a.z; d[3]=a.w;
    d[4]=b.x; d[5]=b.y; d[6]=b.z; d[7]=b.w;
  }
}

// ---------------------------------------------------------------------------
// Kernel 1: EdgeBlock  h_e = relu([cat_e | cat_x[snd] | g] @ W_eb + b_eb)
// Fused: scatter atomicAdd into agg[rcv], edge count, column sums of h_e.
// ---------------------------------------------------------------------------
__global__ __launch_bounds__(256) void edge_block_kernel(
    const float* __restrict__ cat_x, const float* __restrict__ cat_e,
    const int* __restrict__ snd, const int* __restrict__ rcv,
    const float* __restrict__ g, const float* __restrict__ b_eb,
    const v16bf* __restrict__ WebF, float* __restrict__ h_e,
    float* __restrict__ agg, float* __restrict__ cnt,
    float* __restrict__ sum_he) {
  __shared__ v16bf sW[72 * 32];   // 73,728 B: W_eb bf16 fragments
  __shared__ float sSum[128];
  const int tid = threadIdx.x;
  {
    const uint4* s = (const uint4*)WebF;
    uint4* d = (uint4*)sW;
    for (int i = tid; i < 72 * 32 * 2; i += 256) d[i] = s[i];
  }
  if (tid < 128) sSum[tid] = 0.f;
  __syncthreads();

  const int lane  = tid & 31;
  const int wave  = tid >> 5;
  const int khalf = lane >> 4;
  const int mrow  = lane & 15;
  const int gw    = blockIdx.x * 8 + wave;
  const int gstr  = gridDim.x * 8;

  float colsum[8];
#pragma unroll
  for (int t = 0; t < 8; ++t) colsum[t] = 0.f;

  for (int tile = gw; tile < N_EDGES / 16; tile += gstr) {
    const int e0 = tile * 16;
    const int e  = e0 + mrow;
    const int s  = snd[e];
    if (lane < 16) atomicAdd(&cnt[rcv[e]], 1.0f);
    int rcv8[8];
#pragma unroll
    for (int i = 0; i < 8; ++i) rcv8[i] = rcv[e0 + khalf * 8 + i];

    const float* ce_row = cat_e + (size_t)e * 96;
    const float* cx_row = cat_x + (size_t)s * 128;

    const int nt = tile + gstr;
    if (nt < N_EDGES / 16) {  // gfx1250 global_prefetch_b8
      __builtin_prefetch(cat_e + (size_t)(nt * 16 + mrow) * 96, 0, 1);
      __builtin_prefetch(snd + nt * 16 + mrow, 0, 1);
    }

    v8f acc[8] = {};
#pragma unroll
    for (int ss = 0; ss < 9; ++ss) {
      const int kb = ss * 32 + khalf * 8;
      float t0[8], t1[8];
      load8_edge(ce_row, cx_row, g, kb, t0);
      load8_edge(ce_row, cx_row, g, kb + 16, t1);
      v16bf a;
#pragma unroll
      for (int i = 0; i < 8; ++i) { a[i] = (bf16_t)t0[i]; a[8 + i] = (bf16_t)t1[i]; }
#pragma unroll
      for (int t = 0; t < 8; ++t)
        acc[t] = wmma_bf16(a, sW[(t * 9 + ss) * 32 + lane], acc[t]);
    }
#pragma unroll
    for (int t = 0; t < 8; ++t) {
      const int n = t * 16 + mrow;
      const float bias = b_eb[n];
#pragma unroll
      for (int r = 0; r < 8; ++r) {
        float v = fmaxf(acc[t][r] + bias, 0.f);
        const int row = e0 + khalf * 8 + r;
        h_e[(size_t)row * 128 + n] = v;
        colsum[t] += v;
        atomicAdd(&agg[(size_t)rcv8[r] * 128 + n], v);
      }
    }
  }
#pragma unroll
  for (int t = 0; t < 8; ++t) atomicAdd(&sSum[t * 16 + mrow], colsum[t]);
  __syncthreads();
  if (tid < 128) atomicAdd(&sum_he[tid], sSum[tid]);
}

// ---------------------------------------------------------------------------
// Kernel 2: NodeBlock  h_v = relu([agg/cnt | cat_x | g] @ W_nb + b_nb)
// h_v stored as bf16 (feeds nb2 directly). Fused column sums of h_v.
// ---------------------------------------------------------------------------
__global__ __launch_bounds__(256) void node_block_kernel(
    const float* __restrict__ cat_x, const float* __restrict__ g,
    const float* __restrict__ b_nb, const v16bf* __restrict__ WnbF,
    const float* __restrict__ agg, const float* __restrict__ cnt,
    bf16_t* __restrict__ hv, float* __restrict__ sum_hv) {
  __shared__ v16bf sW[80 * 32];   // 81,920 B
  __shared__ float sSum[128];
  const int tid = threadIdx.x;
  {
    const uint4* s = (const uint4*)WnbF;
    uint4* d = (uint4*)sW;
    for (int i = tid; i < 80 * 32 * 2; i += 256) d[i] = s[i];
  }
  if (tid < 128) sSum[tid] = 0.f;
  __syncthreads();

  const int lane = tid & 31, wave = tid >> 5;
  const int khalf = lane >> 4, mrow = lane & 15;
  const int gw = blockIdx.x * 8 + wave, gstr = gridDim.x * 8;

  float colsum[8];
#pragma unroll
  for (int t = 0; t < 8; ++t) colsum[t] = 0.f;

  for (int tile = gw; tile < N_NODES / 16; tile += gstr) {
    const int v0 = tile * 16;
    const int vr = v0 + mrow;
    const float inv = 1.f / fmaxf(cnt[vr], 1.f);
    const float* ag_row = agg + (size_t)vr * 128;
    const float* cx_row = cat_x + (size_t)vr * 128;

    v8f acc[8] = {};
#pragma unroll
    for (int ss = 0; ss < 10; ++ss) {
      const int kb = ss * 32 + khalf * 8;
      float t0[8], t1[8];
      load8_node(ag_row, cx_row, g, inv, kb, t0);
      load8_node(ag_row, cx_row, g, inv, kb + 16, t1);
      v16bf a;
#pragma unroll
      for (int i = 0; i < 8; ++i) { a[i] = (bf16_t)t0[i]; a[8 + i] = (bf16_t)t1[i]; }
#pragma unroll
      for (int t = 0; t < 8; ++t)
        acc[t] = wmma_bf16(a, sW[(t * 10 + ss) * 32 + lane], acc[t]);
    }
#pragma unroll
    for (int t = 0; t < 8; ++t) {
      const int n = t * 16 + mrow;
      const float bias = b_nb[n];
#pragma unroll
      for (int r = 0; r < 8; ++r) {
        float v = fmaxf(acc[t][r] + bias, 0.f);
        hv[(size_t)(v0 + khalf * 8 + r) * 128 + n] = (bf16_t)v;
        colsum[t] += v;
      }
    }
  }
#pragma unroll
  for (int t = 0; t < 8; ++t) atomicAdd(&sSum[t * 16 + mrow], colsum[t]);
  __syncthreads();
  if (tid < 128) atomicAdd(&sum_hv[tid], sSum[tid]);
}

// ---------------------------------------------------------------------------
// Kernel 3: GlobalBlock  u = relu([mean h_e | mean h_v | g] @ W_gb + b_gb)
// ---------------------------------------------------------------------------
__global__ void global_block_kernel(const float* __restrict__ sum_he,
                                    const float* __restrict__ sum_hv,
                                    const float* __restrict__ g,
                                    const float* __restrict__ W_gb,
                                    const float* __restrict__ b_gb,
                                    float* __restrict__ u_tmp,
                                    float* __restrict__ u_out) {
  const int j = threadIdx.x;
  if (j >= 64) return;
  float acc = b_gb[j];
  const float invE = 1.f / (float)N_EDGES, invN = 1.f / (float)N_NODES;
  for (int k = 0; k < 128; ++k) acc += sum_he[k] * invE * W_gb[(size_t)k * 64 + j];
  for (int k = 0; k < 128; ++k) acc += sum_hv[k] * invN * W_gb[(size_t)(128 + k) * 64 + j];
  for (int k = 0; k < 64; ++k)  acc += g[k] * W_gb[(size_t)(256 + k) * 64 + j];
  acc = fmaxf(acc, 0.f);
  u_tmp[j] = acc;
  u_out[j] = acc;
}

// ---------------------------------------------------------------------------
// Kernel 4: nb2  h_v2 = relu([agg/cnt | h_v | u] @ W_nb2a + b) @ W_nb2b + b2
// Two chained WMMA GEMMs; intermediate transposed through LDS.
// ---------------------------------------------------------------------------
__global__ __launch_bounds__(256) void nb2_kernel(
    const float* __restrict__ u, const float* __restrict__ b_nb2a,
    const float* __restrict__ b_nb2b, const v16bf* __restrict__ WaF,
    const v16bf* __restrict__ WbF, const float* __restrict__ agg,
    const float* __restrict__ cnt, const bf16_t* __restrict__ hv,
    float* __restrict__ out2) {
  __shared__ v16bf sWa[80 * 32];                                  // 81,920 B
  __shared__ v16bf sWb[16 * 32];                                  // 16,384 B
  __shared__ __attribute__((aligned(16))) bf16_t sT[8][16][128];  // 32,768 B
  const int tid = threadIdx.x;
  {
    const uint4* s = (const uint4*)WaF;
    uint4* d = (uint4*)sWa;
    for (int i = tid; i < 80 * 32 * 2; i += 256) d[i] = s[i];
    const uint4* s2 = (const uint4*)WbF;
    uint4* d2 = (uint4*)sWb;
    for (int i = tid; i < 16 * 32 * 2; i += 256) d2[i] = s2[i];
  }
  __syncthreads();

  const int lane = tid & 31, wave = tid >> 5;
  const int khalf = lane >> 4, mrow = lane & 15;
  const int gw = blockIdx.x * 8 + wave, gstr = gridDim.x * 8;

  for (int tile = gw; tile < N_NODES / 16; tile += gstr) {
    const int v0 = tile * 16;
    const int vr = v0 + mrow;
    const float inv = 1.f / fmaxf(cnt[vr], 1.f);
    const float* ag_row = agg + (size_t)vr * 128;
    const bf16_t* hv_row = hv + (size_t)vr * 128;

    v8f acc[8] = {};
#pragma unroll
    for (int ss = 0; ss < 10; ++ss) {
      v16bf a;
#pragma unroll
      for (int half = 0; half < 2; ++half) {
        const int k0 = ss * 32 + khalf * 8 + half * 16;
        if (k0 < 128) {
          float4 x = *(const float4*)(ag_row + k0);
          float4 y = *(const float4*)(ag_row + k0 + 4);
          a[half*8+0]=(bf16_t)(x.x*inv); a[half*8+1]=(bf16_t)(x.y*inv);
          a[half*8+2]=(bf16_t)(x.z*inv); a[half*8+3]=(bf16_t)(x.w*inv);
          a[half*8+4]=(bf16_t)(y.x*inv); a[half*8+5]=(bf16_t)(y.y*inv);
          a[half*8+6]=(bf16_t)(y.z*inv); a[half*8+7]=(bf16_t)(y.w*inv);
        } else if (k0 < 256) {
          v8bf r = *(const v8bf*)(hv_row + (k0 - 128));
#pragma unroll
          for (int i = 0; i < 8; ++i) a[half * 8 + i] = r[i];
        } else {
          const float* src = u + (k0 - 256);
          float4 x = *(const float4*)src;
          float4 y = *(const float4*)(src + 4);
          a[half*8+0]=(bf16_t)x.x; a[half*8+1]=(bf16_t)x.y;
          a[half*8+2]=(bf16_t)x.z; a[half*8+3]=(bf16_t)x.w;
          a[half*8+4]=(bf16_t)y.x; a[half*8+5]=(bf16_t)y.y;
          a[half*8+6]=(bf16_t)y.z; a[half*8+7]=(bf16_t)y.w;
        }
      }
#pragma unroll
      for (int t = 0; t < 8; ++t)
        acc[t] = wmma_bf16(a, sWa[(t * 10 + ss) * 32 + lane], acc[t]);
    }
    // bias + relu, write D-layout -> row-major LDS tile (transpose for GEMM2)
#pragma unroll
    for (int t = 0; t < 8; ++t) {
      const int n = t * 16 + mrow;
      const float bias = b_nb2a[n];
#pragma unroll
      for (int r = 0; r < 8; ++r)
        sT[wave][khalf * 8 + r][n] = (bf16_t)fmaxf(acc[t][r] + bias, 0.f);
    }
    asm volatile("s_wait_dscnt 0" ::: "memory");  // intra-wave LDS transpose

    v8f acc2[4] = {};
#pragma unroll
    for (int ss = 0; ss < 4; ++ss) {
      const int kb = ss * 32 + khalf * 8;
      v8bf r1 = *(const v8bf*)&sT[wave][mrow][kb];
      v8bf r2 = *(const v8bf*)&sT[wave][mrow][kb + 16];
      v16bf a2;
#pragma unroll
      for (int i = 0; i < 8; ++i) { a2[i] = r1[i]; a2[8 + i] = r2[i]; }
#pragma unroll
      for (int t = 0; t < 4; ++t)
        acc2[t] = wmma_bf16(a2, sWb[(t * 4 + ss) * 32 + lane], acc2[t]);
    }
#pragma unroll
    for (int t = 0; t < 4; ++t) {
      const int n = t * 16 + mrow;
      const float bias = b_nb2b[n];
#pragma unroll
      for (int r = 0; r < 8; ++r)
        out2[(size_t)(v0 + khalf * 8 + r) * 64 + n] = acc2[t][r] + bias;
    }
  }
}

// ---------------------------------------------------------------------------
// Launch
// ---------------------------------------------------------------------------
extern "C" void kernel_launch(void* const* d_in, const int* in_sizes, int n_in,
                              void* d_out, int out_size, void* d_ws,
                              size_t ws_size, hipStream_t stream) {
  (void)in_sizes; (void)n_in; (void)out_size; (void)ws_size;
  const float* cat_x  = (const float*)d_in[0];
  const float* cat_e  = (const float*)d_in[1];
  const int*   snd    = (const int*)d_in[2];
  const int*   rcv    = (const int*)d_in[3];
  const float* g      = (const float*)d_in[4];
  const float* W_eb   = (const float*)d_in[5];
  const float* b_eb   = (const float*)d_in[6];
  const float* W_nb   = (const float*)d_in[7];
  const float* b_nb   = (const float*)d_in[8];
  const float* W_gb   = (const float*)d_in[9];
  const float* b_gb   = (const float*)d_in[10];
  const float* W_nb2a = (const float*)d_in[11];
  const float* b_nb2a = (const float*)d_in[12];
  const float* W_nb2b = (const float*)d_in[13];
  const float* b_nb2b = (const float*)d_in[14];

  char* ws = (char*)d_ws;
  float* wsf     = (float*)d_ws;
  float* agg     = wsf + AGG_F;
  float* cnt     = wsf + CNT_F;
  float* sum_he  = wsf + SUMHE_F;
  float* sum_hv  = wsf + SUMHV_F;
  float* u_tmp   = wsf + UTMP_F;
  bf16_t* hv     = (bf16_t*)(ws + HV_B);
  bf16_t* WebF   = (bf16_t*)(ws + WEB_B);
  bf16_t* WnbF   = (bf16_t*)(ws + WNB_B);
  bf16_t* WaF    = (bf16_t*)(ws + WA_B);
  bf16_t* WbF    = (bf16_t*)(ws + WB_B);

  float* out   = (float*)d_out;
  float* h_e   = out + OUT_HE;
  float* hv2   = out + OUT_HV2;
  float* u_out = out + OUT_U;

  zero_f32<<<2048, 256, 0, stream>>>(wsf, ZCOUNT);
  pack_w<<<64, 256, 0, stream>>>(W_eb,   WebF, 9,  8, 128);
  pack_w<<<64, 256, 0, stream>>>(W_nb,   WnbF, 10, 8, 128);
  pack_w<<<64, 256, 0, stream>>>(W_nb2a, WaF,  10, 8, 128);
  pack_w<<<64, 256, 0, stream>>>(W_nb2b, WbF,  4,  4, 64);

  edge_block_kernel<<<1024, 256, 0, stream>>>(cat_x, cat_e, snd, rcv, g, b_eb,
                                              (const v16bf*)WebF, h_e, agg,
                                              cnt, sum_he);
  node_block_kernel<<<512, 256, 0, stream>>>(cat_x, g, b_nb,
                                             (const v16bf*)WnbF, agg, cnt, hv,
                                             sum_hv);
  global_block_kernel<<<1, 64, 0, stream>>>(sum_he, sum_hv, g, W_gb, b_gb,
                                            u_tmp, u_out);
  nb2_kernel<<<512, 256, 0, stream>>>(u_tmp, b_nb2a, b_nb2b,
                                      (const v16bf*)WaF, (const v16bf*)WbF,
                                      agg, cnt, hv, hv2);
}
